// CoordAttentionV2_29824252903435
// MI455X (gfx1250) — matrix operand order repaired and checked by reading, not compiled, not run on hardware
//
#include <hip/hip_runtime.h>

#define B_ 2
#define N_ 2048
#define M_ 2048
#define FEAT 512
#define IN_DIM_ 515
#define KP 544            // 515 padded to multiple of 32
#define HF 8
#define H_ 9
#define DH 64
#define INNER_FEAT 512
#define INNER_TOTAL 576
#define SCALE_ 0.125f     // 64^-0.5

typedef __attribute__((ext_vector_type(16))) __bf16 v16bf;
typedef __attribute__((ext_vector_type(8))) float v8f;

union FragA { v16bf v; unsigned int u[8]; };
union Acc8 { v8f v; float f[8]; };

#if __has_builtin(__builtin_amdgcn_tensor_load_to_lds)
#define HAVE_TDM 1
#else
#define HAVE_TDM 0
#endif

#if HAVE_TDM
typedef __attribute__((ext_vector_type(4))) unsigned int v4u_t;
typedef __attribute__((ext_vector_type(8))) int v8i_t;
typedef __attribute__((ext_vector_type(4))) int v4i_t;

__device__ __forceinline__ unsigned lds_addr_of(const void* p) {
  return (unsigned)(unsigned long long)(const __attribute__((address_space(3))) void*)p;
}

// 2-D tile DMA: tile_w elements per row, tile_h rows, row stride in elements,
// element size = 2 bytes. Builds the D# per CDNA5 ISA §8.3/§8.4.
__device__ __forceinline__ void tdm_load_2d(unsigned lds, const void* gptr,
                                            unsigned tile_w, unsigned tile_h,
                                            unsigned stride_elems,
                                            unsigned tensor_w) {
  unsigned long long ga = (unsigned long long)(size_t)gptr;
  v4u_t g0;
  g0.x = 1u;                                               // count = 1 valid D#
  g0.y = lds;                                              // lds_addr
  g0.z = (unsigned)ga;                                     // global_addr[31:0]
  g0.w = (unsigned)((ga >> 32) & 0x1FFFFFFu) | (2u << 30); // addr[56:32] | type=2
  v8i_t g1;
  g1[0] = (int)(1u << 16);                                 // data_size = 2 bytes
  g1[1] = (int)((tensor_w & 0xFFFFu) << 16);               // tensor_dim0 lo
  g1[2] = (int)((tensor_w >> 16) | ((tile_h & 0xFFFFu) << 16)); // dim0 hi | dim1 lo
  g1[3] = (int)((tile_h >> 16) | (tile_w << 16));          // dim1 hi | tile_dim0
  g1[4] = (int)(tile_h & 0xFFFFu);                         // tile_dim1 (tile_dim2=0)
  g1[5] = (int)stride_elems;                               // tensor_dim0_stride lo
  g1[6] = 0;
  g1[7] = 0;
  v4i_t z4 = {0, 0, 0, 0};
#if __clang_major__ >= 23
  v8i_t z8 = {0, 0, 0, 0, 0, 0, 0, 0};
  __builtin_amdgcn_tensor_load_to_lds(g0, g1, z4, z4, z8, 0);
#else
  __builtin_amdgcn_tensor_load_to_lds(g0, g1, z4, z4, 0);
#endif
}
#endif  // HAVE_TDM

__device__ __forceinline__ unsigned short f2bf(float x) {
  return __builtin_bit_cast(unsigned short, (__bf16)x);  // native cvt, RNE
}

__device__ __forceinline__ v8f wmma_bf16(const FragA& a, const FragA& b, v8f c) {
  return __builtin_amdgcn_wmma_f32_16x16x32_bf16(false, a.v, false, b.v,
                                                 (short)0, c, false, false);
}

// A-matrix fragment (16x32 tile, row-major in LDS).
// lane L holds row rbase + L%16; dword j -> k = (j<4 ? 2j : 16+2(j-4)) + 8*(L/16)
__device__ __forceinline__ FragA load_fragA(const unsigned short* lds, int rbase,
                                            int stride, int koff, int lane) {
  FragA f;
  const unsigned short* p = lds + (size_t)(rbase + (lane & 15)) * stride + koff;
  int kb = (lane >> 4) << 3;
#pragma unroll
  for (int j = 0; j < 8; ++j) {
    int k = (j < 4 ? 2 * j : 16 + 2 * (j - 4)) + kb;
    f.u[j] = *(const unsigned int*)(p + k);
  }
  return f;
}

// B-matrix fragment (32x16 k-by-n tile; LDS holds it transposed as [n][k]).
// lane L holds col cbase + L%16; dword j -> k = (j<4 ? 2j : 16+2(j-4)) + 8*(L/16)
__device__ __forceinline__ FragA load_fragB(const unsigned short* lds, int cbase,
                                            int stride, int koff, int lane) {
  FragA f;
  const unsigned short* p = lds + (size_t)(cbase + (lane & 15)) * stride + koff;
  int kb = (lane >> 4) << 3;
#pragma unroll
  for (int j = 0; j < 8; ++j) f.u[j] = *(const unsigned int*)(p + kb + (j < 4 ? 2 * j : 16 + 2 * (j - 4)));
  return f;
}

// ---------------------------------------------------------------------------
// Pack kernels
// ---------------------------------------------------------------------------
__global__ void pack_posed_kernel(const float* __restrict__ feat,
                                  const float* __restrict__ coord,
                                  unsigned short* __restrict__ dst, int rows) {
  int i = blockIdx.x * blockDim.x + threadIdx.x;
  int total = rows * KP;
  if (i >= total) return;
  int r = i / KP, c = i % KP;
  float v = 0.0f;
  if (c < FEAT) v = feat[(size_t)r * FEAT + c];
  else if (c < IN_DIM_) v = coord[(size_t)r * 3 + (c - FEAT)];
  dst[i] = f2bf(v);
}

// Weights are packed COLUMN-major: dst[col * kPad + k]  (B-tile layout)
__global__ void pack_weight_kernel(const float* __restrict__ src,
                                   unsigned short* __restrict__ dst,
                                   int inDim, int kPad, int cols, float scale) {
  int i = blockIdx.x * blockDim.x + threadIdx.x;
  int total = kPad * cols;
  if (i >= total) return;
  int r = i / cols, c = i % cols;
  float v = (r < inDim) ? src[(size_t)r * cols + c] * scale : 0.0f;
  dst[(size_t)c * kPad + r] = f2bf(v);
}

__global__ void pack_coord_kernel(const float* __restrict__ cx,
                                  const float* __restrict__ cy,
                                  const float* __restrict__ cscale,
                                  unsigned short* __restrict__ Qc,
                                  unsigned short* __restrict__ Kc) {
  int i = blockIdx.x * blockDim.x + threadIdx.x;
  int total = B_ * N_ * DH;
  if (i >= total) return;
  int r = i / DH, d = i % DH;
  float cs = cscale[0];
  Qc[i] = f2bf((d < 3) ? cx[(size_t)r * 3 + d] * cs : 0.0f);
  Kc[i] = f2bf((d < 3) ? cy[(size_t)r * 3 + d] : 0.0f);
}

// ---------------------------------------------------------------------------
// bf16 WMMA GEMM: C[rows x cols] = A[rows x K] * Bw^T, Bw stored [cols][K].
// 256x64 block tile, 8 waves (32x64 each -> 8 WMMAs/wave/K-step).
// Both tiles via TDM into double-buffered LDS, 1 barrier per K-step.
// OUT_MODE: 0 = bf16 row-major, 1 = f32 row-major,
//           2 = bf16 transposed to vt[((b*H + h)*DH + d)*M + m] (b128 stores)
// ---------------------------------------------------------------------------
template <int OUT_MODE>
__global__ __launch_bounds__(256) void gemm_bf16_kernel(
    const unsigned short* __restrict__ A, const unsigned short* __restrict__ Bw,
    unsigned short* __restrict__ Cb, float* __restrict__ Cf, int K, int cols) {
  __shared__ unsigned short As[2][256 * 32];
  __shared__ unsigned short Bt[2][64 * 32];  // [col][k]

  int tid = threadIdx.x;
  int lane = tid & 31, w = tid >> 5;
  int row0 = blockIdx.y * 256;
  int c0 = blockIdx.x * 64;
  int T = K >> 5;
  int bc = tid >> 2, bq = (tid & 3) * 8;  // B fallback: col, k-quarter

  Acc8 acc[2][4];
#pragma unroll
  for (int mt = 0; mt < 2; ++mt)
#pragma unroll
    for (int nt = 0; nt < 4; ++nt)
#pragma unroll
      for (int g = 0; g < 8; ++g) acc[mt][nt].f[g] = 0.0f;

  // ---- preload K-step 0 ----
  {
#if HAVE_TDM
    if (w == 0) {
      tdm_load_2d(lds_addr_of(&As[0][0]), A + (size_t)row0 * K, 32, 256,
                  (unsigned)K, (unsigned)K);
      tdm_load_2d(lds_addr_of(&Bt[0][0]), Bw + (size_t)c0 * K, 32, 64,
                  (unsigned)K, (unsigned)K);
      __builtin_amdgcn_s_wait_tensorcnt(0);
    }
#else
    {
      const uint4* src = (const uint4*)(A + (size_t)(row0 + tid) * K);
      uint4* dst = (uint4*)(&As[0][tid * 32]);
      dst[0] = src[0]; dst[1] = src[1]; dst[2] = src[2]; dst[3] = src[3];
    }
    *(uint4*)(&Bt[0][bc * 32 + bq]) =
        *(const uint4*)(Bw + (size_t)(c0 + bc) * K + bq);
#endif
    __syncthreads();
  }

  int cur = 0;
  for (int it = 0; it < T; ++it) {
    int nxt = cur ^ 1;
    bool has_next = (it + 1 < T);
    int k0n = (it + 1) << 5;

    // stage next tiles (DMA / registers) before compute
#if HAVE_TDM
    if (has_next && w == 0) {
      tdm_load_2d(lds_addr_of(&As[nxt][0]), A + (size_t)row0 * K + k0n, 32, 256,
                  (unsigned)K, (unsigned)K);
      tdm_load_2d(lds_addr_of(&Bt[nxt][0]), Bw + (size_t)c0 * K + k0n, 32, 64,
                  (unsigned)K, (unsigned)K);
    }
#else
    uint4 a4[4], b4;
    if (has_next) {
      const uint4* src = (const uint4*)(A + (size_t)(row0 + tid) * K + k0n);
      a4[0] = src[0]; a4[1] = src[1]; a4[2] = src[2]; a4[3] = src[3];
      b4 = *(const uint4*)(Bw + (size_t)(c0 + bc) * K + k0n + bq);
    }
#endif

    // compute on current tiles
    FragA af0 = load_fragA(As[cur], w * 32, 32, 0, lane);
    FragA af1 = load_fragA(As[cur], w * 32 + 16, 32, 0, lane);
#pragma unroll
    for (int nt = 0; nt < 4; ++nt) {
      FragA bfr = load_fragB(Bt[cur], nt * 16, 32, 0, lane);
      acc[0][nt].v = wmma_bf16(af0, bfr, acc[0][nt].v);
      acc[1][nt].v = wmma_bf16(af1, bfr, acc[1][nt].v);
    }

    // commit staged tiles
    if (has_next) {
#if !HAVE_TDM
      uint4* dst = (uint4*)(&As[nxt][tid * 32]);
      dst[0] = a4[0]; dst[1] = a4[1]; dst[2] = a4[2]; dst[3] = a4[3];
      *(uint4*)(&Bt[nxt][bc * 32 + bq]) = b4;
#else
      if (w == 0) __builtin_amdgcn_s_wait_tensorcnt(0);
#endif
    }
    __syncthreads();
    cur = nxt;
  }

#pragma unroll
  for (int mt = 0; mt < 2; ++mt)
#pragma unroll
    for (int nt = 0; nt < 4; ++nt) {
      if (OUT_MODE == 2) {
        // lane holds 8 consecutive m for one (b,h,d): one b128 store
        int mg = row0 + w * 32 + mt * 16 + ((lane >> 4) << 3);
        int bb = mg >> 11;          // / M_
        int mm = mg & (M_ - 1);
        int c = c0 + nt * 16 + (lane & 15);
        int hh = c >> 6, dd = c & 63;
        uint4 o4;
        o4.x = (unsigned)f2bf(acc[mt][nt].f[0]) |
               ((unsigned)f2bf(acc[mt][nt].f[1]) << 16);
        o4.y = (unsigned)f2bf(acc[mt][nt].f[2]) |
               ((unsigned)f2bf(acc[mt][nt].f[3]) << 16);
        o4.z = (unsigned)f2bf(acc[mt][nt].f[4]) |
               ((unsigned)f2bf(acc[mt][nt].f[5]) << 16);
        o4.w = (unsigned)f2bf(acc[mt][nt].f[6]) |
               ((unsigned)f2bf(acc[mt][nt].f[7]) << 16);
        *(uint4*)(Cb + ((((size_t)bb * H_ + hh) * DH + dd) * M_ + mm)) = o4;
      } else {
#pragma unroll
        for (int g = 0; g < 8; ++g) {
          int r = row0 + w * 32 + mt * 16 + g + ((lane >> 4) << 3);
          int c = c0 + nt * 16 + (lane & 15);
          if (OUT_MODE == 1)
            Cf[(size_t)r * cols + c] = acc[mt][nt].f[g];
          else
            Cb[(size_t)r * cols + c] = f2bf(acc[mt][nt].f[g]);
        }
      }
    }
}

// ---------------------------------------------------------------------------
// Flash attention: one (batch, head) x 64-row Q tile per block, 4 waves.
// K chunks ([key][d]) and pre-transposed V chunks ([d][key]) both arrive via
// TDM into double-buffered LDS; 1 barrier per chunk.
// ---------------------------------------------------------------------------
__global__ __launch_bounds__(128) void attn_kernel(
    const unsigned short* __restrict__ qbuf, const unsigned short* __restrict__ kbuf,
    const unsigned short* __restrict__ vt, const unsigned short* __restrict__ Qc,
    const unsigned short* __restrict__ Kc, unsigned short* __restrict__ Obuf) {
  __shared__ unsigned short Qs[64 * 64];
  __shared__ unsigned short Ks[2][64 * 64];   // [key][d]
  __shared__ unsigned short Vs[2][64 * 64];   // [d][key]
  __shared__ unsigned short Ps[4][16 * 64];   // per-wave P tile

  int tid = threadIdx.x;
  int lane = tid & 31, w = tid >> 5;
  int bh = blockIdx.y;
  int b = bh / H_, h = bh % H_;
  int n0 = blockIdx.x * 64;

  const unsigned short *qp, *kp;
  int qstride, kstride;
  if (h < HF) {
    qp = qbuf + (size_t)b * N_ * INNER_FEAT + h * DH;
    kp = kbuf + (size_t)b * M_ * INNER_FEAT + h * DH;
    qstride = INNER_FEAT;
    kstride = INNER_FEAT;
  } else {  // coordinate head: padded 3-dim coords
    qp = Qc + (size_t)b * N_ * DH;
    kp = Kc + (size_t)b * M_ * DH;
    qstride = DH;
    kstride = DH;
  }
  const unsigned short* vtp = vt + ((size_t)b * H_ + h) * DH * M_;

  int kr = tid >> 1, kseg = (tid & 1) * 32;   // 64 rows x 2 halves

  // ---- Q tile + chunk-0 preload ----
  {
    const uint4* src = (const uint4*)(qp + (size_t)(n0 + kr) * qstride + kseg);
    uint4* dst = (uint4*)(Qs + kr * 64 + kseg);
    dst[0] = src[0]; dst[1] = src[1]; dst[2] = src[2]; dst[3] = src[3];
  }
#if HAVE_TDM
  if (w == 0) {
    tdm_load_2d(lds_addr_of(&Ks[0][0]), kp, 64, 64, (unsigned)kstride,
                (unsigned)kstride);
    tdm_load_2d(lds_addr_of(&Vs[0][0]), vtp, 64, 64, (unsigned)M_, (unsigned)M_);
    __builtin_amdgcn_s_wait_tensorcnt(0);
  }
#else
  {
    const uint4* src = (const uint4*)(kp + (size_t)kr * kstride + kseg);
    uint4* dst = (uint4*)(&Ks[0][kr * 64 + kseg]);
    dst[0] = src[0]; dst[1] = src[1]; dst[2] = src[2]; dst[3] = src[3];
    const uint4* vs = (const uint4*)(vtp + (size_t)kr * M_ + kseg);
    uint4* vd = (uint4*)(&Vs[0][kr * 64 + kseg]);
    vd[0] = vs[0]; vd[1] = vs[1]; vd[2] = vs[2]; vd[3] = vs[3];
  }
#endif
  __syncthreads();

  FragA qf[2];
  qf[0] = load_fragA(Qs, w * 16, 64, 0, lane);
  qf[1] = load_fragA(Qs, w * 16, 64, 32, lane);

  float mrow[8], lrow[8];
#pragma unroll
  for (int g = 0; g < 8; ++g) { mrow[g] = -3.0e38f; lrow[g] = 0.0f; }
  Acc8 o[4];
#pragma unroll
  for (int nt = 0; nt < 4; ++nt)
#pragma unroll
    for (int g = 0; g < 8; ++g) o[nt].f[g] = 0.0f;

  const float LOG2E = 1.4426950408889634f;
  const int T = M_ / 64;
  int cur = 0;

  for (int mc = 0; mc < T; ++mc) {
    int nxt = cur ^ 1;
    bool has_next = (mc + 1 < T);
    int m0n = (mc + 1) * 64;

    // stage next chunk before compute
#if HAVE_TDM
    if (has_next && w == 0) {
      tdm_load_2d(lds_addr_of(&Ks[nxt][0]), kp + (size_t)m0n * kstride, 64, 64,
                  (unsigned)kstride, (unsigned)kstride);
      tdm_load_2d(lds_addr_of(&Vs[nxt][0]), vtp + m0n, 64, 64, (unsigned)M_,
                  (unsigned)M_);
    }
#else
    uint4 kreg[4], vreg[4];
    if (has_next) {
      const uint4* ks = (const uint4*)(kp + (size_t)(m0n + kr) * kstride + kseg);
      kreg[0] = ks[0]; kreg[1] = ks[1]; kreg[2] = ks[2]; kreg[3] = ks[3];
      const uint4* vs = (const uint4*)(vtp + (size_t)kr * M_ + m0n + kseg);
      vreg[0] = vs[0]; vreg[1] = vs[1]; vreg[2] = vs[2]; vreg[3] = vs[3];
    }
#endif

    // S = Q * K^T  (16 rows x 64 keys per wave)
    Acc8 sacc[4];
#pragma unroll
    for (int nt = 0; nt < 4; ++nt) {
#pragma unroll
      for (int g = 0; g < 8; ++g) sacc[nt].f[g] = 0.0f;
      FragA kf0 = load_fragB(Ks[cur], nt * 16, 64, 0, lane);
      sacc[nt].v = wmma_bf16(qf[0], kf0, sacc[nt].v);
      FragA kf1 = load_fragB(Ks[cur], nt * 16, 64, 32, lane);
      sacc[nt].v = wmma_bf16(qf[1], kf1, sacc[nt].v);
    }

    // online softmax; row g+8*(lane/16) lives across 16 lanes of a half-wave
#pragma unroll
    for (int g = 0; g < 8; ++g) {
      float mx = fmaxf(fmaxf(sacc[0].f[g], sacc[1].f[g]),
                       fmaxf(sacc[2].f[g], sacc[3].f[g]));
      mx = fmaxf(mx, __shfl_xor(mx, 1, 32));
      mx = fmaxf(mx, __shfl_xor(mx, 2, 32));
      mx = fmaxf(mx, __shfl_xor(mx, 4, 32));
      mx = fmaxf(mx, __shfl_xor(mx, 8, 32));
      float newm = fmaxf(mrow[g], mx);
      float alpha = exp2f((mrow[g] - newm) * LOG2E);
      mrow[g] = newm;
      float rs = 0.0f;
#pragma unroll
      for (int nt = 0; nt < 4; ++nt) {
        float pv = exp2f((sacc[nt].f[g] - newm) * LOG2E);
        sacc[nt].f[g] = pv;
        rs += pv;
      }
      rs += __shfl_xor(rs, 1, 32);
      rs += __shfl_xor(rs, 2, 32);
      rs += __shfl_xor(rs, 4, 32);
      rs += __shfl_xor(rs, 8, 32);
      lrow[g] = lrow[g] * alpha + rs;
#pragma unroll
      for (int nt = 0; nt < 4; ++nt) o[nt].f[g] *= alpha;
    }

    // stage P (bf16) into per-wave LDS in A-tile layout
#pragma unroll
    for (int nt = 0; nt < 4; ++nt)
#pragma unroll
      for (int g = 0; g < 8; ++g) {
        int r = g + ((lane >> 4) << 3);
        int c = nt * 16 + (lane & 15);
        Ps[w][r * 64 + c] = f2bf(sacc[nt].f[g]);
      }

    FragA pf0 = load_fragA(Ps[w], 0, 64, 0, lane);
    FragA pf1 = load_fragA(Ps[w], 0, 64, 32, lane);
#pragma unroll
    for (int nt = 0; nt < 4; ++nt) {
      FragA vf0 = load_fragB(Vs[cur], nt * 16, 64, 0, lane);
      o[nt].v = wmma_bf16(pf0, vf0, o[nt].v);
      FragA vf1 = load_fragB(Vs[cur], nt * 16, 64, 32, lane);
      o[nt].v = wmma_bf16(pf1, vf1, o[nt].v);
    }

    // commit staged chunk
    if (has_next) {
#if !HAVE_TDM
      uint4* kd = (uint4*)(&Ks[nxt][kr * 64 + kseg]);
      kd[0] = kreg[0]; kd[1] = kreg[1]; kd[2] = kreg[2]; kd[3] = kreg[3];
      uint4* vd = (uint4*)(&Vs[nxt][kr * 64 + kseg]);
      vd[0] = vreg[0]; vd[1] = vreg[1]; vd[2] = vreg[2]; vd[3] = vreg[3];
#else
      if (w == 0) __builtin_amdgcn_s_wait_tensorcnt(0);
#endif
    }
    __syncthreads();
    cur = nxt;
  }

#pragma unroll
  for (int nt = 0; nt < 4; ++nt)
#pragma unroll
    for (int g = 0; g < 8; ++g) {
      int r = g + ((lane >> 4) << 3);
      float val = o[nt].f[g] * __builtin_amdgcn_rcpf(lrow[g]);
      int gr = n0 + w * 16 + r;
      int gc = h * DH + nt * 16 + (lane & 15);
      Obuf[(size_t)(b * N_ + gr) * INNER_TOTAL + gc] = f2bf(val);
    }
}

// ---------------------------------------------------------------------------
extern "C" void kernel_launch(void* const* d_in, const int* in_sizes, int n_in,
                              void* d_out, int out_size, void* d_ws, size_t ws_size,
                              hipStream_t stream) {
  (void)in_sizes; (void)n_in; (void)out_size; (void)ws_size;
  const float* x  = (const float*)d_in[0];
  const float* y  = (const float*)d_in[1];
  const float* cx = (const float*)d_in[2];
  const float* cy = (const float*)d_in[3];
  const float* Wq = (const float*)d_in[4];
  const float* Wk = (const float*)d_in[5];
  const float* Wv = (const float*)d_in[6];
  const float* Wo = (const float*)d_in[7];
  const float* cs = (const float*)d_in[8];
  float* out = (float*)d_out;

  char* p = (char*)d_ws;
  auto take = [&](size_t bytes) -> unsigned short* {
    unsigned short* r = (unsigned short*)p;
    p += (bytes + 255) & ~(size_t)255;
    return r;
  };
  const int rowsX = B_ * N_;  // 4096 (== B_*M_)
  unsigned short* posedX = take((size_t)rowsX * KP * 2);
  unsigned short* posedY = take((size_t)rowsX * KP * 2);
  unsigned short* WqP = take((size_t)KP * INNER_FEAT * 2);
  unsigned short* WkP = take((size_t)KP * INNER_FEAT * 2);
  unsigned short* WvP = take((size_t)KP * INNER_TOTAL * 2);
  unsigned short* WoP = take((size_t)INNER_TOTAL * INNER_FEAT * 2);
  unsigned short* qb  = take((size_t)rowsX * INNER_FEAT * 2);
  unsigned short* kb  = take((size_t)rowsX * INNER_FEAT * 2);
  unsigned short* vt  = take((size_t)B_ * H_ * DH * M_ * 2);
  unsigned short* Qc  = take((size_t)rowsX * DH * 2);
  unsigned short* Kc  = take((size_t)rowsX * DH * 2);
  unsigned short* Ob  = take((size_t)rowsX * INNER_TOTAL * 2);

  const int tpb = 256;
  int tot;
  tot = rowsX * KP;
  pack_posed_kernel<<<(tot + tpb - 1) / tpb, tpb, 0, stream>>>(x, cx, posedX, rowsX);
  pack_posed_kernel<<<(tot + tpb - 1) / tpb, tpb, 0, stream>>>(y, cy, posedY, rowsX);
  tot = KP * INNER_FEAT;
  pack_weight_kernel<<<(tot + tpb - 1) / tpb, tpb, 0, stream>>>(Wq, WqP, IN_DIM_, KP, INNER_FEAT, SCALE_);
  pack_weight_kernel<<<(tot + tpb - 1) / tpb, tpb, 0, stream>>>(Wk, WkP, IN_DIM_, KP, INNER_FEAT, 1.0f);
  tot = KP * INNER_TOTAL;
  pack_weight_kernel<<<(tot + tpb - 1) / tpb, tpb, 0, stream>>>(Wv, WvP, IN_DIM_, KP, INNER_TOTAL, 1.0f);
  tot = INNER_TOTAL * INNER_FEAT;
  pack_weight_kernel<<<(tot + tpb - 1) / tpb, tpb, 0, stream>>>(Wo, WoP, INNER_TOTAL, INNER_TOTAL, INNER_FEAT, 1.0f);
  tot = rowsX * DH;
  pack_coord_kernel<<<(tot + tpb - 1) / tpb, tpb, 0, stream>>>(cx, cy, cs, Qc, Kc);

  dim3 gq(INNER_FEAT / 64, rowsX / 256);
  gemm_bf16_kernel<0><<<gq, 256, 0, stream>>>(posedX, WqP, qb, (float*)nullptr, KP, INNER_FEAT);
  gemm_bf16_kernel<0><<<gq, 256, 0, stream>>>(posedY, WkP, kb, (float*)nullptr, KP, INNER_FEAT);
  dim3 gv(INNER_TOTAL / 64, rowsX / 256);
  gemm_bf16_kernel<2><<<gv, 256, 0, stream>>>(posedY, WvP, vt, (float*)nullptr, KP, INNER_TOTAL);

  dim3 ga(N_ / 64, B_ * H_);
  attn_kernel<<<ga, 128, 0, stream>>>(qb, kb, vt, Qc, Kc, Ob);

  dim3 go(INNER_FEAT / 64, rowsX / 256);
  gemm_bf16_kernel<1><<<go, 256, 0, stream>>>(Ob, WoP, (unsigned short*)nullptr, out, INNER_TOTAL, INNER_FEAT);
}
